// DeepFM_42863773614392
// MI455X (gfx1250) — compile-verified, matrix operand order checked by reading, not run on hardware
//
#include <hip/hip_runtime.h>

// ---------------- problem constants ----------------
#define B_BATCH  16384
#define F_FIELDS 26
#define VOCAB    100000
#define EMB_D    32
#define N_DENSE  13
#define K0PAD    864          // 26*32 + 13 = 845 -> padded to 27*32
#define H1DIM    1024
#define H2DIM    512
#define H3DIM    256

typedef __attribute__((ext_vector_type(16))) _Float16 v16h;
typedef __attribute__((ext_vector_type(8)))  _Float16 h8;
typedef __attribute__((ext_vector_type(8)))  float    v8f;

union Frag16 { v16h v; h8 h[2]; };

// ---------------------------------------------------------------
// Weight convert + transpose: W[K][N] f32 -> Wt[N][Kpad] f16 (zero pad K..Kpad)
// ---------------------------------------------------------------
__global__ __launch_bounds__(256) void convert_wt(const float* __restrict__ W,
                                                  _Float16* __restrict__ Wt,
                                                  int K, int N, int Kpad) {
    int i = blockIdx.x * 256 + threadIdx.x;
    if (i >= N * Kpad) return;
    int n = i / Kpad;
    int k = i - n * Kpad;
    float v = (k < K) ? W[(size_t)k * N + n] : 0.0f;
    Wt[i] = (_Float16)v;
}

// ---------------------------------------------------------------
// Fused embedding gather + linear + FM + DNN-input build.
// One wave (32 lanes = embedding dim) per sample; 8 samples / block.
// ---------------------------------------------------------------
__global__ __launch_bounds__(256) void embed_fm_kernel(
    const int*   __restrict__ xs,     // [B, 26]
    const float* __restrict__ xd,     // [B, 13]
    const float* __restrict__ emb,    // [26, V, 32]
    const float* __restrict__ lin,    // [26, V]
    _Float16*    __restrict__ A0,     // [B, 864] f16
    float*       __restrict__ partial // [B] = linear_logit + fm_logit
) {
    const int wave = threadIdx.x >> 5;
    const int lane = threadIdx.x & 31;
    const int b = blockIdx.x * 8 + wave;

    const int* idx = xs + (size_t)b * F_FIELDS;
    float s = 0.0f, sq = 0.0f;

    #pragma unroll 4
    for (int f = 0; f < F_FIELDS; ++f) {
        int ix = idx[f];                                    // uniform across wave
        float e = emb[((size_t)f * VOCAB + (size_t)ix) * EMB_D + lane];
        s  += e;
        sq += e * e;
        A0[(size_t)b * K0PAD + f * EMB_D + lane] = (_Float16)e;
    }

    float linv = 0.0f;
    if (lane < F_FIELDS)
        linv = lin[(size_t)lane * VOCAB + idx[lane]];

    // per-lane contribution to 0.5*(||s||^2 - sum ||e||^2) plus linear term
    float t = 0.5f * (s * s - sq) + linv;
    #pragma unroll
    for (int o = 16; o > 0; o >>= 1) t += __shfl_xor(t, o, 32);
    if (lane == 0) partial[b] = t;

    // dense features + zero padding (cols 832..863)
    float dv = (lane < N_DENSE) ? xd[(size_t)b * N_DENSE + lane] : 0.0f;
    A0[(size_t)b * K0PAD + F_FIELDS * EMB_D + lane] = (_Float16)dv;
}

// ---------------------------------------------------------------
// f16 WMMA GEMM:  C[M,N] = act(A[M,K] * Bt[N,K]^T + bias)
// Block tile 128 x (NB*64), 8 waves (2 along M x 4 along N),
// each wave computes 64 x (NB*16): 4 x NB fragments -> 4*NB WMMAs
// per 32-wide k-step against 4+NB LDS fragment loads.
// NB=4 -> 128x256 block tile; NB=2 -> 128x128.
// ---------------------------------------------------------------
template<int NB>
__global__ __launch_bounds__(256) void gemm_bias_act(
    const _Float16* __restrict__ A,   // [M, K]
    const _Float16* __restrict__ Bt,  // [N, K]  (pre-transposed weights)
    const float*    __restrict__ bias,// [N]
    _Float16*       __restrict__ C,   // [M, N]
    int M, int N, int K, int relu)
{
    constexpr int BN = NB * 64;        // block N extent
    __shared__ _Float16 sA[128][40];   // 32 used + 8 pad (bank spread)
    __shared__ _Float16 sB[BN][40];

    const int tid  = threadIdx.x;
    const int wave = tid >> 5;
    const int lane = tid & 31;
    const int r    = lane & 15;        // M index in A frag / N index in B frag
    const int hi   = lane >> 4;        // lane-group select per ISA layout
    const int m0   = blockIdx.y * 128;
    const int n0   = blockIdx.x * BN;
    const int wm   = (wave >> 2) * 64;        // 2 waves along M
    const int wn   = (wave & 3) * (NB * 16);  // 4 waves along N

    v8f acc[4][NB];
    #pragma unroll
    for (int i = 0; i < 4; ++i)
        #pragma unroll
        for (int j = 0; j < NB; ++j) {
            v8f z = {0.f,0.f,0.f,0.f,0.f,0.f,0.f,0.f};
            acc[i][j] = z;
        }

    for (int k0 = 0; k0 < K; k0 += 32) {
        // stage A[128x32] and Bt[BNx32] tiles (16B chunks, fully coalesced)
        #pragma unroll
        for (int i = 0; i < 2; ++i) {
            int chunk = tid + i * 256;       // 0..511
            int row   = chunk >> 2;          // 0..127
            int col   = (chunk & 3) * 8;     // 0,8,16,24
            *(h8*)&sA[row][col] = *(const h8*)&A[(size_t)(m0 + row) * K + k0 + col];
        }
        #pragma unroll
        for (int i = 0; i < NB; ++i) {
            int chunk = tid + i * 256;       // 0 .. BN*4-1
            int row   = chunk >> 2;          // 0 .. BN-1
            int col   = (chunk & 3) * 8;
            *(h8*)&sB[row][col] = *(const h8*)&Bt[(size_t)(n0 + row) * K + k0 + col];
        }
        __syncthreads();

        // per-lane fragment loads: halves 0..7 -> k = 8*hi + e, halves 8..15 -> k = 16 + 8*hi + e
        Frag16 af[4], bf[NB];
        #pragma unroll
        for (int fm = 0; fm < 4; ++fm) {
            int row = wm + fm * 16 + r;
            af[fm].h[0] = *(const h8*)&sA[row][8 * hi];
            af[fm].h[1] = *(const h8*)&sA[row][16 + 8 * hi];
        }
        #pragma unroll
        for (int fn = 0; fn < NB; ++fn) {
            int row = wn + fn * 16 + r;
            bf[fn].h[0] = *(const h8*)&sB[row][8 * hi];
            bf[fn].h[1] = *(const h8*)&sB[row][16 + 8 * hi];
        }

        #pragma unroll
        for (int fm = 0; fm < 4; ++fm)
            #pragma unroll
            for (int fn = 0; fn < NB; ++fn)
                acc[fm][fn] = __builtin_amdgcn_wmma_f32_16x16x32_f16(
                    false, af[fm].v, false, bf[fn].v,
                    (short)0, acc[fm][fn], false, false);
        __syncthreads();
    }

    // writeback: lane holds (m = i + 8*hi, n = lane&15) per C/D VGPR layout
    #pragma unroll
    for (int fn = 0; fn < NB; ++fn) {
        int n  = n0 + wn + fn * 16 + r;
        float bv = bias[n];
        #pragma unroll
        for (int fm = 0; fm < 4; ++fm) {
            #pragma unroll
            for (int i = 0; i < 8; ++i) {
                int m = m0 + wm + fm * 16 + i + 8 * hi;
                float v = acc[fm][fn][i] + bv;
                if (relu) v = v > 0.0f ? v : 0.0f;
                C[(size_t)m * N + n] = (_Float16)v;
            }
        }
    }
}

// ---------------------------------------------------------------
// Output head: dnn_logit = H3 . Wout + bout ; sigmoid(partial + dnn_logit)
// One wave per sample.
// ---------------------------------------------------------------
__global__ __launch_bounds__(256) void final_kernel(
    const _Float16* __restrict__ H3,    // [B, 256]
    const float*    __restrict__ Wout,  // [256]
    const float*    __restrict__ bout,  // [1]
    const float*    __restrict__ partial,
    float*          __restrict__ out)
{
    const int wave = threadIdx.x >> 5;
    const int lane = threadIdx.x & 31;
    const int b = blockIdx.x * 8 + wave;

    float acc = 0.0f;
    #pragma unroll
    for (int j = lane; j < H3DIM; j += 32)
        acc += (float)H3[(size_t)b * H3DIM + j] * Wout[j];
    #pragma unroll
    for (int o = 16; o > 0; o >>= 1) acc += __shfl_xor(acc, o, 32);

    if (lane == 0) {
        float z = acc + bout[0] + partial[b];
        out[b] = 1.0f / (1.0f + __expf(-z));
    }
}

// ---------------------------------------------------------------
extern "C" void kernel_launch(void* const* d_in, const int* in_sizes, int n_in,
                              void* d_out, int out_size, void* d_ws, size_t ws_size,
                              hipStream_t stream) {
    const int*   x_sparse = (const int*)  d_in[0];
    const float* x_dense  = (const float*)d_in[1];
    const float* emb      = (const float*)d_in[2];
    const float* lin      = (const float*)d_in[3];
    const float* W0   = (const float*)d_in[4];
    const float* b0   = (const float*)d_in[5];
    const float* W1   = (const float*)d_in[6];
    const float* b1   = (const float*)d_in[7];
    const float* W2   = (const float*)d_in[8];
    const float* b2   = (const float*)d_in[9];
    const float* Wout = (const float*)d_in[10];
    const float* bout = (const float*)d_in[11];
    float* out = (float*)d_out;

    char*  ws  = (char*)d_ws;
    size_t off = 0;
    auto alloc = [&](size_t bytes) -> void* {
        void* p = ws + off;
        off += (bytes + 255) & ~(size_t)255;
        return p;
    };

    float*    partial = (float*)   alloc((size_t)B_BATCH * 4);
    _Float16* A0      = (_Float16*)alloc((size_t)B_BATCH * K0PAD * 2);
    _Float16* H1      = (_Float16*)alloc((size_t)B_BATCH * H1DIM * 2);
    _Float16* H2      = (_Float16*)alloc((size_t)B_BATCH * H2DIM * 2);
    _Float16* H3      = (_Float16*)alloc((size_t)B_BATCH * H3DIM * 2);
    _Float16* W0t     = (_Float16*)alloc((size_t)H1DIM * K0PAD * 2);
    _Float16* W1t     = (_Float16*)alloc((size_t)H2DIM * H1DIM * 2);
    _Float16* W2t     = (_Float16*)alloc((size_t)H3DIM * H2DIM * 2);

    // 1) weight conversion / transpose (f32 -> f16, [N][Kpad])
    {
        int n0 = H1DIM * K0PAD;
        convert_wt<<<(n0 + 255) / 256, 256, 0, stream>>>(W0, W0t, F_FIELDS * EMB_D + N_DENSE, H1DIM, K0PAD);
        int n1 = H2DIM * H1DIM;
        convert_wt<<<(n1 + 255) / 256, 256, 0, stream>>>(W1, W1t, H1DIM, H2DIM, H1DIM);
        int n2 = H3DIM * H2DIM;
        convert_wt<<<(n2 + 255) / 256, 256, 0, stream>>>(W2, W2t, H2DIM, H3DIM, H2DIM);
    }

    // 2) fused gather + FM + linear + DNN-input build
    embed_fm_kernel<<<B_BATCH / 8, 256, 0, stream>>>(x_sparse, x_dense, emb, lin, A0, partial);

    // 3) WMMA GEMM stack (f16 in, f32 accumulate, bias+ReLU fused)
    //    layers 0/1: 128x256 block tile (64x64 wave tile)
    //    layer  2  : 128x128 block tile (64x32 wave tile) for occupancy at N=256
    gemm_bias_act<4><<<dim3(H1DIM / 256, B_BATCH / 128), 256, 0, stream>>>(A0, W0t, b0, H1, B_BATCH, H1DIM, K0PAD, 1);
    gemm_bias_act<4><<<dim3(H2DIM / 256, B_BATCH / 128), 256, 0, stream>>>(H1, W1t, b1, H2, B_BATCH, H2DIM, H1DIM, 1);
    gemm_bias_act<2><<<dim3(H3DIM / 128, B_BATCH / 128), 256, 0, stream>>>(H2, W2t, b2, H3, B_BATCH, H3DIM, H2DIM, 1);

    // 4) output head + sigmoid
    final_kernel<<<B_BATCH / 8, 256, 0, stream>>>(H3, Wout, bout, partial, out);
}
